// RLSAttnCogVideoXAttnProcessor_34325378629690
// MI455X (gfx1250) — compile-verified, matrix operand order checked by reading, not run on hardware
//
#include <hip/hip_runtime.h>
#include <hip/hip_bf16.h>

#define S_TOT  2274
#define T_TXT  226
#define HEADS  30
#define HDIM   64
#define DMODEL 1920
#define SPAD   2304
#define EPSV   1e-5f

typedef __bf16 bf16_t;
typedef __attribute__((ext_vector_type(16))) __bf16 v16bf;
typedef __attribute__((ext_vector_type(8)))  float  v8f;
typedef __attribute__((ext_vector_type(4)))  int    i32x4;

union FragBF {
  v16bf v;
  uint4 q[2];
};

// ---- CDNA5 async global->LDS copy (ASYNCcnt path), with safe fallback ------
#if defined(__has_builtin)
#  if __has_builtin(__builtin_amdgcn_global_load_async_to_lds_b128)
#    define HAVE_ASYNC_LDS 1
#  endif
#endif

#ifdef HAVE_ASYNC_LDS
typedef __attribute__((address_space(1))) i32x4 Gi32x4;  // global int4
typedef __attribute__((address_space(3))) i32x4 Li32x4;  // LDS int4
#  if __has_builtin(__builtin_amdgcn_s_wait_asynccnt)
#    define WAIT_ASYNC() __builtin_amdgcn_s_wait_asynccnt(0)
#  else
#    define WAIT_ASYNC() asm volatile("s_wait_asynccnt 0x0" ::: "memory")
#  endif
#  define ASYNC_CP16(dst_lds, src_glb)                                        \
     __builtin_amdgcn_global_load_async_to_lds_b128(                          \
         (Gi32x4*)(void*)(src_glb), (Li32x4*)(void*)(dst_lds), 0, 0)
#else
#  define WAIT_ASYNC() ((void)0)
#  define ASYNC_CP16(dst_lds, src_glb)                                        \
     (*(uint4*)(dst_lds) = *(const uint4*)(src_glb))
#endif

// ---------------------------------------------------------------------------
// Elementwise: concat [encoder ; hidden] -> bf16 x
// ---------------------------------------------------------------------------
__global__ __launch_bounds__(256) void concat_x_kernel(
    const float* __restrict__ ehs, const float* __restrict__ hs,
    bf16_t* __restrict__ xb)
{
  const int i = blockIdx.x * blockDim.x + threadIdx.x;
  if (i >= S_TOT * DMODEL) return;
  const int s = i / DMODEL;
  const int c = i - s * DMODEL;
  float v = (s < T_TXT) ? ehs[(size_t)s * DMODEL + c]
                        : hs[(size_t)(s - T_TXT) * DMODEL + c];
  xb[i] = (bf16_t)v;
}

__global__ __launch_bounds__(256) void cvt_bf16_kernel(
    const float* __restrict__ src, bf16_t* __restrict__ dst, int n)
{
  const int i = blockIdx.x * blockDim.x + threadIdx.x;
  if (i < n) dst[i] = (bf16_t)src[i];
}

// ---------------------------------------------------------------------------
// WMMA GEMM: C[M,N] = A[M,K] * Bw[N,K]^T + bias  (A, Bw bf16, C f32)
// Block tile 128(M) x 128(N), 8 waves; wave tile 32x64 = 2x4 WMMA frags
// (8 v_wmma per k-step). Tiles staged via async global->LDS when available.
// remapT>0: output row s<remapT goes to the encoder section after the
// hidden section (CogVideoX split-output layout).
// ---------------------------------------------------------------------------
__global__ __launch_bounds__(256) void gemm_bf16_wmma(
    const bf16_t* __restrict__ A, const bf16_t* __restrict__ Bw,
    const float* __restrict__ bias, float* __restrict__ C,
    int M, int N, int K, int remapT)
{
  __shared__ bf16_t lA[128 * 40];   // 32 cols + 8 pad (rows stay 16B aligned)
  __shared__ bf16_t lB[128 * 40];

  const int tid  = threadIdx.x;
  const int lane = tid & 31;
  const int wave = tid >> 5;
  const int half = lane >> 4;
  const int lr   = lane & 15;
  const int tileM = blockIdx.y * 128;
  const int tileN = blockIdx.x * 128;
  const int wm = (wave >> 1) * 32;   // wave's M sub-tile origin (0..96)
  const int wn = (wave & 1) * 64;    // wave's N sub-tile origin (0 or 64)

  v8f acc[2][4] = {};

  for (int k0 = 0; k0 < K; k0 += 32) {
    __syncthreads();
    // A tile: 128 rows x 32 bf16 (512 b128 chunks, 2 per thread).
    // Partial last M-tile: clamp row (dup data, rows >= M never stored).
#pragma unroll
    for (int i = 0; i < 2; ++i) {
      const int idx = tid + i * 256;
      const int row = idx >> 2;
      const int ch  = (idx & 3) * 8;
      int gr = tileM + row;
      if (gr >= M) gr = M - 1;
      ASYNC_CP16(&lA[row * 40 + ch], A + (size_t)gr * K + k0 + ch);
    }
    // B tile: 128 rows x 32 bf16 (N is a multiple of 128 -> no clamp)
#pragma unroll
    for (int i = 0; i < 2; ++i) {
      const int idx = tid + i * 256;
      const int row = idx >> 2;
      const int ch  = (idx & 3) * 8;
      ASYNC_CP16(&lB[row * 40 + ch], Bw + (size_t)(tileN + row) * K + k0 + ch);
    }
    if (k0 + 32 < K) {
      __builtin_prefetch(A  + (size_t)(tileM + (tid & 127)) * K + k0 + 32, 0, 1);
      __builtin_prefetch(Bw + (size_t)(tileN + (tid & 127)) * K + k0 + 32, 0, 1);
    }
    WAIT_ASYNC();
    __syncthreads();

    // A fragments (CDNA5 16-bit A layout: lane=row, VGPR0-3 K=half*8..+7,
    // VGPR4-7 K=half*8+16..+23)
    FragBF af[2];
#pragma unroll
    for (int a2 = 0; a2 < 2; ++a2) {
      const bf16_t* p = &lA[(wm + a2 * 16 + lr) * 40 + half * 8];
      af[a2].q[0] = *(const uint4*)p;
      af[a2].q[1] = *(const uint4*)(p + 16);
    }
#pragma unroll
    for (int t = 0; t < 4; ++t) {
      FragBF bfr;
      const bf16_t* p = &lB[(wn + t * 16 + lr) * 40 + half * 8];
      bfr.q[0] = *(const uint4*)p;
      bfr.q[1] = *(const uint4*)(p + 16);
      acc[0][t] = __builtin_amdgcn_wmma_f32_16x16x32_bf16(
          false, af[0].v, false, bfr.v, (short)0, acc[0][t], false, false);
      acc[1][t] = __builtin_amdgcn_wmma_f32_16x16x32_bf16(
          false, af[1].v, false, bfr.v, (short)0, acc[1][t], false, false);
    }
  }

  // Epilogue: C layout lane=col (lr), VGPR r = row r + 8*half
#pragma unroll
  for (int a2 = 0; a2 < 2; ++a2) {
#pragma unroll
    for (int t = 0; t < 4; ++t) {
      const int col = tileN + wn + t * 16 + lr;
      const float bv = bias[col];
#pragma unroll
      for (int r = 0; r < 8; ++r) {
        const int row = tileM + wm + a2 * 16 + half * 8 + r;
        if (row < M) {
          size_t o;
          if (remapT > 0)
            o = (row < remapT)
                  ? ((size_t)(M - remapT) * N + (size_t)row * N + col)
                  : ((size_t)(row - remapT) * N + col);
          else
            o = (size_t)row * N + col;
          C[o] = acc[a2][t][r] + bv;
        }
      }
    }
  }
}

// ---------------------------------------------------------------------------
// Fused per-head LayerNorm (+RoPE for image tokens), pack to bf16 [H][S][64]
// One wave32 per (q/k, head, token) row; 2 elems/lane, shuffle reductions.
// ---------------------------------------------------------------------------
__global__ __launch_bounds__(256) void ln_rope_pack(
    const float* __restrict__ qf, const float* __restrict__ kf,
    const float* __restrict__ nqw, const float* __restrict__ nqb,
    const float* __restrict__ nkw, const float* __restrict__ nkb,
    const float* __restrict__ rc,  const float* __restrict__ rsn,
    bf16_t* __restrict__ qb, bf16_t* __restrict__ kb)
{
  const int lane = threadIdx.x & 31;
  const int gw = (int)((blockIdx.x * blockDim.x + threadIdx.x) >> 5);
  const int total = 2 * HEADS * S_TOT;
  if (gw >= total) return;
  const int isk = (gw >= HEADS * S_TOT) ? 1 : 0;
  const int rem = isk ? gw - HEADS * S_TOT : gw;
  const int h = rem % HEADS;
  const int s = rem / HEADS;
  const float* src = isk ? kf : qf;
  const float* wv  = isk ? nkw : nqw;
  const float* bv  = isk ? nkb : nqb;
  bf16_t* dst = isk ? kb : qb;

  const size_t base = (size_t)s * DMODEL + h * HDIM;
  const int d0 = lane * 2;
  float x0 = src[base + d0];
  float x1 = src[base + d0 + 1];

  float sum = x0 + x1;
#pragma unroll
  for (int m = 1; m < 32; m <<= 1) sum += __shfl_xor(sum, m, 32);
  const float mu = sum * (1.0f / 64.0f);
  float e0 = x0 - mu, e1 = x1 - mu;
  float vs = e0 * e0 + e1 * e1;
#pragma unroll
  for (int m = 1; m < 32; m <<= 1) vs += __shfl_xor(vs, m, 32);
  const float inv = rsqrtf(vs * (1.0f / 64.0f) + EPSV);

  float y0 = e0 * inv * wv[d0]     + bv[d0];
  float y1 = e1 * inv * wv[d0 + 1] + bv[d0 + 1];

  if (s >= T_TXT) { // RoPE on image tokens; lane holds an (even,odd) pair
    const int p = s - T_TXT;
    const float c0 = rc[p * HDIM + d0],  c1 = rc[p * HDIM + d0 + 1];
    const float s0 = rsn[p * HDIM + d0], s1 = rsn[p * HDIM + d0 + 1];
    const float z0 = y0 * c0 - y1 * s0;  // rot[2i]   = -x[2i+1]
    const float z1 = y1 * c1 + y0 * s1;  // rot[2i+1] =  x[2i]
    y0 = z0; y1 = z1;
  }
  const size_t ob = ((size_t)h * S_TOT + s) * HDIM + d0;
  dst[ob]     = (bf16_t)y0;
  dst[ob + 1] = (bf16_t)y1;
}

// V -> bf16, transposed [H][64][SPAD] (zero-padded keys) so PV B-fragments
// are contiguous 16B chunks over the key dimension.
__global__ __launch_bounds__(256) void v_pack_T(
    const float* __restrict__ vf, bf16_t* __restrict__ vT)
{
  const int i = blockIdx.x * blockDim.x + threadIdx.x;
  const int n = HEADS * HDIM * SPAD;
  if (i >= n) return;
  const int sp = i % SPAD;
  const int hd = i / SPAD;
  const int d  = hd % HDIM;
  const int h  = hd / HDIM;
  float v = 0.0f;
  if (sp < S_TOT) v = vf[(size_t)sp * DMODEL + h * HDIM + d];
  vT[i] = (bf16_t)v;
}

// ---------------------------------------------------------------------------
// Flash attention: grid (qtile, head), 4 waves x 16 queries, stream 32-key
// tiles. Scores and PV both via v_wmma_f32_16x16x32_bf16; P transposed
// C-layout -> A-layout through per-wave LDS (same-wave DS ordering).
// ---------------------------------------------------------------------------
__global__ __launch_bounds__(128) void attn_fwd(
    const bf16_t* __restrict__ Qh, const bf16_t* __restrict__ Kh,
    const bf16_t* __restrict__ VT, bf16_t* __restrict__ Ob)
{
  __shared__ bf16_t lp[4][16 * 40];   // per-wave P staging (16x32 + pad)

  const int h    = blockIdx.y;
  const int q0   = blockIdx.x * 64;
  const int wave = threadIdx.x >> 5;
  const int lane = threadIdx.x & 31;
  const int half = lane >> 4;
  const int lr   = lane & 15;
  const int qbase = q0 + wave * 16;

  const bf16_t* Qp = Qh + (size_t)h * S_TOT * HDIM;
  const bf16_t* Kp = Kh + (size_t)h * S_TOT * HDIM;
  const bf16_t* Vp = VT + (size_t)h * HDIM * SPAD;

  // Q fragments (A layout): 2 chunks of K=32 over HDIM=64
  int qrow = qbase + lr;
  if (qrow >= S_TOT) qrow = S_TOT - 1;
  FragBF qfr[2];
#pragma unroll
  for (int c = 0; c < 2; ++c) {
    const bf16_t* p = Qp + (size_t)qrow * HDIM + c * 32 + half * 8;
    qfr[c].q[0] = *(const uint4*)p;
    qfr[c].q[1] = *(const uint4*)(p + 16);
  }

  v8f o[4] = {};
  float mrow[8], lrow[8];
#pragma unroll
  for (int r = 0; r < 8; ++r) { mrow[r] = -1e30f; lrow[r] = 0.0f; }
  const float scale = 0.125f; // 1/sqrt(64)

  const int nkt = (S_TOT + 31) / 32;
  for (int kt = 0; kt < nkt; ++kt) {
    const int n0 = kt * 32;

    // ---- scores: two 16x16 tiles over this 32-key strip
    v8f sc[2] = {};
#pragma unroll
    for (int t = 0; t < 2; ++t) {
      int key = n0 + t * 16 + lr;
      if (key >= S_TOT) key = S_TOT - 1;
#pragma unroll
      for (int c = 0; c < 2; ++c) {
        FragBF kfr; // B layout: lane=key column, chunks over head-dim
        const bf16_t* p = Kp + (size_t)key * HDIM + c * 32 + half * 8;
        kfr.q[0] = *(const uint4*)p;
        kfr.q[1] = *(const uint4*)(p + 16);
        sc[t] = __builtin_amdgcn_wmma_f32_16x16x32_bf16(
            false, qfr[c].v, false, kfr.v, (short)0, sc[t], false, false);
      }
    }

    // ---- online softmax (rows live as VGPR index r + 8*half; cols = lanes)
    const bool mask0 = (n0 + lr)      >= S_TOT;
    const bool mask1 = (n0 + 16 + lr) >= S_TOT;
    float p0[8], p1[8];
#pragma unroll
    for (int r = 0; r < 8; ++r) {
      float a = mask0 ? -1e30f : sc[0][r] * scale;
      float b = mask1 ? -1e30f : sc[1][r] * scale;
      float v = fmaxf(a, b);
      v = fmaxf(v, __shfl_xor(v, 1, 32));
      v = fmaxf(v, __shfl_xor(v, 2, 32));
      v = fmaxf(v, __shfl_xor(v, 4, 32));
      v = fmaxf(v, __shfl_xor(v, 8, 32));
      const float mn = fmaxf(mrow[r], v);
      const float alpha = __expf(mrow[r] - mn);
      mrow[r] = mn;
      const float e0 = __expf(a - mn);
      const float e1 = __expf(b - mn);
      p0[r] = e0; p1[r] = e1;
      float rs = e0 + e1;
      rs += __shfl_xor(rs, 1, 32);
      rs += __shfl_xor(rs, 2, 32);
      rs += __shfl_xor(rs, 4, 32);
      rs += __shfl_xor(rs, 8, 32);
      lrow[r] = lrow[r] * alpha + rs;
      o[0][r] *= alpha; o[1][r] *= alpha; o[2][r] *= alpha; o[3][r] *= alpha;
    }

    // ---- transpose P (C layout -> A layout) through per-wave LDS
    bf16_t* myP = lp[wave];
#pragma unroll
    for (int r = 0; r < 8; ++r) {
      myP[(r + 8 * half) * 40 + lr]      = (bf16_t)p0[r];
      myP[(r + 8 * half) * 40 + 16 + lr] = (bf16_t)p1[r];
    }
    FragBF pfr;
    {
      const bf16_t* p = &myP[lr * 40 + half * 8];
      pfr.q[0] = *(const uint4*)p;
      pfr.q[1] = *(const uint4*)(p + 16);
    }

    // ---- O += P(16x32) x V(32x64): four output tiles, V^T rows contiguous
#pragma unroll
    for (int t = 0; t < 4; ++t) {
      FragBF vfr;
      const bf16_t* p = Vp + (size_t)(t * 16 + lr) * SPAD + n0 + half * 8;
      vfr.q[0] = *(const uint4*)p;
      vfr.q[1] = *(const uint4*)(p + 16);
      o[t] = __builtin_amdgcn_wmma_f32_16x16x32_bf16(
          false, pfr.v, false, vfr.v, (short)0, o[t], false, false);
    }
  }

  // ---- normalize and store bf16 [S][D] token-major for the O projection
#pragma unroll
  for (int t = 0; t < 4; ++t) {
#pragma unroll
    for (int r = 0; r < 8; ++r) {
      const int row = qbase + half * 8 + r;
      if (row < S_TOT) {
        const float val = o[t][r] / lrow[r];
        Ob[(size_t)row * DMODEL + h * HDIM + t * 16 + lr] = (bf16_t)val;
      }
    }
  }
}

// ---------------------------------------------------------------------------
extern "C" void kernel_launch(void* const* d_in, const int* in_sizes, int n_in,
                              void* d_out, int out_size, void* d_ws, size_t ws_size,
                              hipStream_t stream)
{
  const float* hs  = (const float*)d_in[0];
  const float* ehs = (const float*)d_in[1];
  const float* rc  = (const float*)d_in[2];
  const float* rsn = (const float*)d_in[3];
  const float* Wq  = (const float*)d_in[4];
  const float* bq  = (const float*)d_in[5];
  const float* Wk  = (const float*)d_in[6];
  const float* bk  = (const float*)d_in[7];
  const float* Wv  = (const float*)d_in[8];
  const float* bv  = (const float*)d_in[9];
  const float* nqw = (const float*)d_in[10];
  const float* nqb = (const float*)d_in[11];
  const float* nkw = (const float*)d_in[12];
  const float* nkb = (const float*)d_in[13];
  const float* Wo  = (const float*)d_in[14];
  const float* bo  = (const float*)d_in[15];

  char* w = (char*)d_ws;
  size_t off = 0;
  auto alloc = [&](size_t bytes) -> char* {
    char* p = w + off;
    off += (bytes + 255) & ~(size_t)255;
    return p;
  };

  bf16_t* xb  = (bf16_t*)alloc((size_t)S_TOT * DMODEL * 2);
  bf16_t* wqb = (bf16_t*)alloc((size_t)DMODEL * DMODEL * 2);
  bf16_t* wkb = (bf16_t*)alloc((size_t)DMODEL * DMODEL * 2);
  bf16_t* wvb = (bf16_t*)alloc((size_t)DMODEL * DMODEL * 2);
  bf16_t* wob = (bf16_t*)alloc((size_t)DMODEL * DMODEL * 2);
  float*  qf  = (float*)alloc((size_t)S_TOT * DMODEL * 4);
  float*  kf  = (float*)alloc((size_t)S_TOT * DMODEL * 4);
  float*  vf  = (float*)alloc((size_t)S_TOT * DMODEL * 4);
  bf16_t* qhb = (bf16_t*)alloc((size_t)HEADS * S_TOT * HDIM * 2);
  bf16_t* khb = (bf16_t*)alloc((size_t)HEADS * S_TOT * HDIM * 2);
  bf16_t* vtb = (bf16_t*)alloc((size_t)HEADS * HDIM * SPAD * 2);
  bf16_t* ob  = (bf16_t*)alloc((size_t)S_TOT * DMODEL * 2);

  const int nx = S_TOT * DMODEL;
  concat_x_kernel<<<(nx + 255) / 256, 256, 0, stream>>>(ehs, hs, xb);

  const int nw = DMODEL * DMODEL;
  cvt_bf16_kernel<<<(nw + 255) / 256, 256, 0, stream>>>(Wq, wqb, nw);
  cvt_bf16_kernel<<<(nw + 255) / 256, 256, 0, stream>>>(Wk, wkb, nw);
  cvt_bf16_kernel<<<(nw + 255) / 256, 256, 0, stream>>>(Wv, wvb, nw);
  cvt_bf16_kernel<<<(nw + 255) / 256, 256, 0, stream>>>(Wo, wob, nw);

  dim3 gg(DMODEL / 128, (S_TOT + 127) / 128);
  gemm_bf16_wmma<<<gg, 256, 0, stream>>>(xb, wqb, bq, qf, S_TOT, DMODEL, DMODEL, 0);
  gemm_bf16_wmma<<<gg, 256, 0, stream>>>(xb, wkb, bk, kf, S_TOT, DMODEL, DMODEL, 0);
  gemm_bf16_wmma<<<gg, 256, 0, stream>>>(xb, wvb, bv, vf, S_TOT, DMODEL, DMODEL, 0);

  const int nlr = 2 * HEADS * S_TOT; // waves
  ln_rope_pack<<<(nlr * 32 + 255) / 256, 256, 0, stream>>>(
      qf, kf, nqw, nqb, nkw, nkb, rc, rsn, qhb, khb);

  const int nv = HEADS * HDIM * SPAD;
  v_pack_T<<<(nv + 255) / 256, 256, 0, stream>>>(vf, vtb);

  dim3 ga((S_TOT + 63) / 64, HEADS);
  attn_fwd<<<ga, 128, 0, stream>>>(qhb, khb, vtb, ob);

  gemm_bf16_wmma<<<gg, 256, 0, stream>>>(ob, wob, bo, (float*)d_out,
                                         S_TOT, DMODEL, DMODEL, T_TXT);
}
